// EnhancedGNNEncoder_58463094833685
// MI455X (gfx1250) — compile-verified, matrix operand order checked by reading, not run on hardware
//
#include <hip/hip_runtime.h>
#include <hip/hip_bf16.h>

typedef _Float16 half_t;
typedef __attribute__((ext_vector_type(16))) _Float16 v16h;
typedef __attribute__((ext_vector_type(8)))  _Float16 v8h;
typedef __attribute__((ext_vector_type(8)))  float    v8f;

__device__ __forceinline__ float siluf(float x){ return x / (1.0f + __expf(-x)); }
__device__ __forceinline__ float sigmf(float x){ return 1.0f / (1.0f + __expf(-x)); }

// Build a 16-half WMMA operand fragment. For the CDNA5 16-bit operand layout,
// a lane's elements 0..7 are K = k0+8*hi .. +7 (contiguous) and elements 8..15
// are K = k0+16+8*hi .. +7 (contiguous), so two 16B loads suffice.
__device__ __forceinline__ v16h load_frag(const half_t* p){
  v8h a = *(const v8h*)(p);
  v8h b = *(const v8h*)(p + 16);
  v16h r;
#pragma unroll
  for (int i = 0; i < 8; ++i){ r[i] = a[i]; r[i + 8] = b[i]; }
  return r;
}

// 16x16 output tile over K, A row-major [16][aStride] (f16, LDS),
// B pre-transposed weights [n][bStride] (f16, global), D = A*B.
template<int K>
__device__ __forceinline__ v8f wmma_rowcol(const half_t* aBase, int aStride,
                                           const half_t* bBase, int bStride){
  const int lane = threadIdx.x & 31;
  const int idx  = lane & 15;
  const int hi   = lane >> 4;
  const half_t* pa = aBase + idx * aStride + 8 * hi;
  const half_t* pb = bBase + idx * bStride + 8 * hi;
  __builtin_prefetch(pb, 0, 3);
  v8f acc = {};
#pragma unroll
  for (int k0 = 0; k0 < K; k0 += 32){
    v16h af = load_frag(pa + k0);
    v16h bf = load_frag(pb + k0);
    acc = __builtin_amdgcn_wmma_f32_16x16x32_f16(false, af, false, bf,
                                                 (short)0, acc, false, false);
  }
  return acc;
}

// ---------------- prep kernels ----------------

__global__ void transpose_f16_kernel(const float* __restrict__ src,
                                     half_t* __restrict__ dst, int K, int Nn){
  int i = blockIdx.x * blockDim.x + threadIdx.x;
  if (i < K * Nn){
    int n = i / K, k = i % K;
    dst[n * K + k] = (half_t)src[k * Nn + n];
  }
}

__global__ void conv_h_kernel(const float* __restrict__ h,
                              half_t* __restrict__ hh, int n){
  int i = blockIdx.x * blockDim.x + threadIdx.x;
  if (i < n) hh[i] = (half_t)h[i];
}

__global__ void node_prep_kernel(const float* __restrict__ x,
                                 const float* __restrict__ cw,
                                 float* __restrict__ pc,
                                 float* __restrict__ chs, int N){
  int n = blockIdx.x * blockDim.x + threadIdx.x;
  if (n >= N) return;
  int cnt = 0; float s0 = 0.f, s1 = 0.f, s2 = 0.f;
#pragma unroll
  for (int c = 0; c < 14; ++c){
    float w = cw[n * 14 + c];
    if (w != 0.0f){
      ++cnt;
      s0 += x[n * 42 + c * 3 + 0];
      s1 += x[n * 42 + c * 3 + 1];
      s2 += x[n * 42 + c * 3 + 2];
    }
  }
  float inv = cnt > 0 ? 1.0f / (float)cnt : 0.0f;
  pc[n * 3 + 0] = s0 * inv; pc[n * 3 + 1] = s1 * inv; pc[n * 3 + 2] = s2 * inv;
  chs[n] = (float)cnt;
}

__global__ void count_kernel(const int* __restrict__ row, const int* __restrict__ col,
                             float* __restrict__ cr, float* __restrict__ cc, int E){
  int e = blockIdx.x * blockDim.x + threadIdx.x;
  if (e < E){
    atomicAdd(&cr[row[e]], 1.0f);
    atomicAdd(&cc[col[e]], 1.0f);
  }
}

// ---------------- fused edge kernel: 16 edges / block, 8 waves ----------------

__global__ __launch_bounds__(256) void edge_kernel(
    const float* __restrict__ x, const float* __restrict__ cw,
    const float* __restrict__ attr,
    const int* __restrict__ row, const int* __restrict__ col,
    const half_t* __restrict__ hh,
    const half_t* __restrict__ radT, const float* __restrict__ rad_b,
    const half_t* __restrict__ e1T,  const float* __restrict__ e_b1,
    const half_t* __restrict__ e2T,  const float* __restrict__ e_b2,
    const float* __restrict__ att_w, const float* __restrict__ att_b,
    const half_t* __restrict__ c1T,  const float* __restrict__ c_b1,
    const float* __restrict__ c_w2,  const float* __restrict__ c_b2,
    const float* __restrict__ pc,    const float* __restrict__ chsum,
    float* __restrict__ agg, float* __restrict__ xacc, int E)
{
  __shared__ int    sRow[16], sCol[16];
  __shared__ float  sChs[16], sNormSq[16], sGate[16];
  __shared__ float  sPc[16 * 3];
  __shared__ float  sXr[16 * 42];
  __shared__ __align__(16) half_t sRh[16 * 256];
  __shared__ __align__(16) char   sU[31360];   // phase-overlapped region

  // phase-1 views
  float* sXc = (float*)(sU);            // 16*42
  float* sWr = (float*)(sU + 2688);     // 16*14
  float* sWc = (float*)(sU + 3584);     // 16*14
  float* sM  = (float*)(sU + 4480);     // 16*196
  float* sT  = (float*)(sU + 17024);    // 16*224 ([c][b])
  // phase-2 views (phase-1 arrays dead by first write)
  half_t* sA   = (half_t*)(sU);           // 16*384  concat(h_r,h_c,radial) f16
  half_t* sH1  = (half_t*)(sU + 12288);   // 16*128  f16
  float*  sEf  = (float*)(sU + 16384);    // 16*128  f32
  half_t* sEfH = (half_t*)(sU + 24576);   // 16*128  f16
  float*  sCm  = (float*)(sU + 28672);    // 16*14

  const int tid  = threadIdx.x;
  const int lane = tid & 31;
  const int wave = tid >> 5;
  const int e0   = blockIdx.x * 16;

  if (tid < 16){
    int eg = e0 + tid; if (eg >= E) eg = E - 1;
    sRow[tid] = row[eg];
    sCol[tid] = col[eg];
    sNormSq[tid] = 0.0f;
  }
  __syncthreads();
  if (tid < 16) sChs[tid] = chsum[sRow[tid]];
  if (tid >= 64 && tid < 112){
    int i = tid - 64, e = i / 3, d = i % 3;
    sPc[i] = pc[sCol[e] * 3 + d];
  }
  for (int i = tid; i < 16 * 42; i += 256){
    int e = i / 42, r = i % 42;
    sXr[i] = x[sRow[e] * 42 + r];
    sXc[i] = x[sCol[e] * 42 + r];
  }
  for (int i = tid; i < 16 * 14; i += 256){
    int e = i / 14, c = i % 14;
    sWr[i] = cw[sRow[e] * 14 + c];
    sWc[i] = cw[sCol[e] * 14 + c];
  }
  __syncthreads();

  // ---- M[e][c][d] = ||x_r[c]-x_c[d]|| * w_r[c]*w_c[d] ----
  for (int i = tid; i < 16 * 196; i += 256){
    int e = i / 196, p = i % 196, c = p / 14, d = p % 14;
    const float* xr = &sXr[e * 42 + c * 3];
    const float* xc = &sXc[e * 42 + d * 3];
    float dx = xr[0] - xc[0], dy = xr[1] - xc[1], dz = xr[2] - xc[2];
    sM[i] = sqrtf(dx * dx + dy * dy + dz * dz) * sWr[e * 14 + c] * sWc[e * 14 + d];
  }
  __syncthreads();

  // ---- T[e][c][b] = sum_d M[c][d] * Ac[d][b] ----
  for (int i = tid; i < 16 * 224; i += 256){
    int e = i / 224, p = i % 224, c = p / 16, b = p % 16;
    const float* ac = attr + (size_t)sCol[e] * 224 + b;
    const float* m  = &sM[e * 196 + c * 14];
    float s = 0.f;
#pragma unroll
    for (int d = 0; d < 14; ++d) s += m[d] * ac[d * 16];
    sT[e * 224 + c * 16 + b] = s;
  }
  __syncthreads();

  // ---- R[e][a*16+b] = sum_c Ar[c][a] * T[c][b] ; accumulate ||R||^2 ----
  for (int i = tid; i < 16 * 256; i += 256){
    int e = i / 256, p = i % 256, a = p / 16, b = p % 16;
    const float* ar = attr + (size_t)sRow[e] * 224 + a;
    const float* t  = &sT[e * 224 + b];
    float s = 0.f;
#pragma unroll
    for (int c = 0; c < 14; ++c) s += ar[c * 16] * t[c * 16];
    sRh[i] = (half_t)s;
    atomicAdd(&sNormSq[e], s * s);
  }
  __syncthreads();

  // ---- radial = (R @ rad_w + b) / (||R||+1) -> sA[:,256:384]; also copy h ----
  {
    v8f acc = wmma_rowcol<256>(sRh, 256, radT + wave * 16 * 256, 256);
    int nl = lane & 15, hi = lane >> 4, ng = wave * 16 + nl;
    float b = rad_b[ng];
#pragma unroll
    for (int v = 0; v < 8; ++v){
      int m = v + 8 * hi;
      float denom = sqrtf(sNormSq[m]) + 1.0f;
      sA[m * 384 + 256 + ng] = (half_t)((acc[v] + b) / denom);
    }
  }
  for (int i = tid; i < 16 * 256; i += 256){
    int e = i / 256, k = i % 256;
    sA[e * 384 + k] = (k < 128) ? hh[(size_t)sRow[e] * 128 + k]
                                : hh[(size_t)sCol[e] * 128 + (k - 128)];
  }
  __syncthreads();

  // ---- layer1: silu(concat @ e_w1 + b1) ----
  {
    v8f acc = wmma_rowcol<384>(sA, 384, e1T + wave * 16 * 384, 384);
    int nl = lane & 15, hi = lane >> 4, ng = wave * 16 + nl;
    float b = e_b1[ng];
#pragma unroll
    for (int v = 0; v < 8; ++v)
      sH1[(v + 8 * hi) * 128 + ng] = (half_t)siluf(acc[v] + b);
  }
  __syncthreads();

  // ---- layer2: silu(. @ e_w2 + b2) -> sEf (f32) ----
  {
    v8f acc = wmma_rowcol<128>(sH1, 128, e2T + wave * 16 * 128, 128);
    int nl = lane & 15, hi = lane >> 4, ng = wave * 16 + nl;
    float b = e_b2[ng];
#pragma unroll
    for (int v = 0; v < 8; ++v)
      sEf[(v + 8 * hi) * 128 + ng] = siluf(acc[v] + b);
  }
  __syncthreads();

  // ---- attention gate ----
  if (tid < 16){
    float s = 0.f;
#pragma unroll 4
    for (int n = 0; n < 128; ++n) s += sEf[tid * 128 + n] * att_w[n];
    sGate[tid] = sigmf(s + att_b[0]);
  }
  __syncthreads();
  for (int i = tid; i < 16 * 128; i += 256){
    int e = i >> 7;
    float v = sEf[i] * sGate[e];
    sEf[i]  = v;
    sEfH[i] = (half_t)v;
  }
  __syncthreads();

  // ---- scatter ef into agg[col] ----
  for (int i = tid; i < 16 * 128; i += 256){
    int e = i >> 7, n = i & 127;
    if (e0 + e < E) atomicAdd(&agg[(size_t)sCol[e] * 128 + n], sEf[i]);
  }
  __syncthreads();

  // ---- coord hidden: silu(ef @ c_w1 + b) -> sEf (reused, f32) ----
  {
    v8f acc = wmma_rowcol<128>(sEfH, 128, c1T + wave * 16 * 128, 128);
    int nl = lane & 15, hi = lane >> 4, ng = wave * 16 + nl;
    float b = c_b1[ng];
#pragma unroll
    for (int v = 0; v < 8; ++v)
      sEf[(v + 8 * hi) * 128 + ng] = siluf(acc[v] + b);
  }
  __syncthreads();

  // ---- cm = hidden @ c_w2 + b  (128 -> 14) ----
  for (int i = tid; i < 16 * 14; i += 256){
    int e = i / 14, j = i % 14;
    float s = 0.f;
#pragma unroll 4
    for (int k = 0; k < 128; ++k) s += sEf[e * 128 + k] * c_w2[k * 14 + j];
    sCm[i] = s + c_b2[j];
  }
  __syncthreads();

  // ---- RollerPooling + trans scatter into xacc[row] ----
  for (int i = tid; i < 16 * 14; i += 256){
    int e = i / 14, r = i % 14;
    int eg = e0 + e;
    if (eg >= E) continue;
    int chs = (int)sChs[e];
    int ws  = 15 - chs;                 // C - (channel_sum - 1)
    int jend = r + ws; if (jend > 14) jend = 14;
    float s = 0.f;
    for (int j = r; j < jend; ++j) s += sCm[e * 14 + j];
    float pooled = s / (float)ws;
    int nrow = sRow[e];
#pragma unroll
    for (int d = 0; d < 3; ++d){
      float diff = sXr[e * 42 + r * 3 + d] - sPc[e * 3 + d];
      atomicAdd(&xacc[(size_t)nrow * 42 + r * 3 + d], diff * pooled);
    }
  }
}

// ---------------- node update: GEMM + GEMM + residual + LayerNorm ----------------

__global__ __launch_bounds__(256) void node_kernel(
    const float* __restrict__ h, const half_t* __restrict__ hh,
    const float* __restrict__ agg, const float* __restrict__ cc,
    const half_t* __restrict__ n1T, const float* __restrict__ n_b1,
    const half_t* __restrict__ n2T, const float* __restrict__ n_b2,
    const float* __restrict__ ln_g, const float* __restrict__ ln_b,
    float* __restrict__ out, int N)
{
  __shared__ __align__(16) half_t sA[16 * 256];
  __shared__ __align__(16) half_t sH[16 * 128];
  __shared__ float sR[16 * 128];
  __shared__ float sMu[16], sVar[16];

  const int tid = threadIdx.x, lane = tid & 31, wave = tid >> 5;
  const int n0 = blockIdx.x * 16;

  for (int i = tid; i < 16 * 256; i += 256){
    int r = i >> 8, k = i & 255;
    int node = n0 + r; if (node >= N) node = N - 1;
    half_t v;
    if (k < 128) v = hh[(size_t)node * 128 + k];
    else {
      float c = cc[node]; if (c < 1.0f) c = 1.0f;
      v = (half_t)(agg[(size_t)node * 128 + (k - 128)] / c);
    }
    sA[i] = v;
  }
  __syncthreads();
  {
    v8f acc = wmma_rowcol<256>(sA, 256, n1T + wave * 16 * 256, 256);
    int nl = lane & 15, hi = lane >> 4, ng = wave * 16 + nl;
    float b = n_b1[ng];
#pragma unroll
    for (int v = 0; v < 8; ++v)
      sH[(v + 8 * hi) * 128 + ng] = (half_t)siluf(acc[v] + b);
  }
  __syncthreads();
  {
    v8f acc = wmma_rowcol<128>(sH, 128, n2T + wave * 16 * 128, 128);
    int nl = lane & 15, hi = lane >> 4, ng = wave * 16 + nl;
    float b = n_b2[ng];
#pragma unroll
    for (int v = 0; v < 8; ++v){
      int m = v + 8 * hi;
      int node = n0 + m; if (node >= N) node = N - 1;
      sR[m * 128 + ng] = h[(size_t)node * 128 + ng] + acc[v] + b;
    }
  }
  __syncthreads();
  if (tid < 16){
    float mu = 0.f;
    for (int k = 0; k < 128; ++k) mu += sR[tid * 128 + k];
    mu *= (1.0f / 128.0f);
    float var = 0.f;
    for (int k = 0; k < 128; ++k){ float d = sR[tid * 128 + k] - mu; var += d * d; }
    sMu[tid] = mu; sVar[tid] = var * (1.0f / 128.0f);
  }
  __syncthreads();
  for (int i = tid; i < 16 * 128; i += 256){
    int r = i >> 7, k = i & 127;
    int node = n0 + r;
    if (node < N)
      out[(size_t)node * 128 + k] =
          (sR[i] - sMu[r]) * rsqrtf(sVar[r] + 1e-5f) * ln_g[k] + ln_b[k];
  }
}

__global__ void xout_kernel(const float* __restrict__ x, const float* __restrict__ xacc,
                            const float* __restrict__ cr, float* __restrict__ outX, int N){
  int i = blockIdx.x * blockDim.x + threadIdx.x;
  if (i < N * 42){
    float c = cr[i / 42]; if (c < 1.0f) c = 1.0f;
    outX[i] = x[i] + xacc[i] / c;
  }
}

// ---------------- host ----------------

extern "C" void kernel_launch(void* const* d_in, const int* in_sizes, int n_in,
                              void* d_out, int out_size, void* d_ws, size_t ws_size,
                              hipStream_t stream)
{
  const float* h     = (const float*)d_in[0];
  const float* x     = (const float*)d_in[1];
  const float* attr  = (const float*)d_in[2];
  const float* cw    = (const float*)d_in[3];
  const int*   row   = (const int*)d_in[4];
  const int*   col   = (const int*)d_in[5];
  const float* rad_w = (const float*)d_in[6];
  const float* rad_b = (const float*)d_in[7];
  const float* e_w1  = (const float*)d_in[8];
  const float* e_b1  = (const float*)d_in[9];
  const float* e_w2  = (const float*)d_in[10];
  const float* e_b2  = (const float*)d_in[11];
  const float* att_w = (const float*)d_in[12];
  const float* att_b = (const float*)d_in[13];
  const float* c_w1  = (const float*)d_in[14];
  const float* c_b1  = (const float*)d_in[15];
  const float* c_w2  = (const float*)d_in[16];
  const float* c_b2  = (const float*)d_in[17];
  const float* n_w1  = (const float*)d_in[18];
  const float* n_b1  = (const float*)d_in[19];
  const float* n_w2  = (const float*)d_in[20];
  const float* n_b2  = (const float*)d_in[21];
  const float* ln_g  = (const float*)d_in[22];
  const float* ln_b  = (const float*)d_in[23];

  const int N = in_sizes[0] / 128;
  const int E = in_sizes[4];

  char* base = (char*)d_ws;
  size_t off = 0;
  auto alloc = [&](size_t bytes) -> void* {
    off = (off + 255) & ~(size_t)255;
    void* p = base + off;
    off += bytes;
    return p;
  };

  half_t* hh   = (half_t*)alloc((size_t)N * 128 * 2);
  half_t* radT = (half_t*)alloc((size_t)128 * 256 * 2);
  half_t* e1T  = (half_t*)alloc((size_t)128 * 384 * 2);
  half_t* e2T  = (half_t*)alloc((size_t)128 * 128 * 2);
  half_t* c1T  = (half_t*)alloc((size_t)128 * 128 * 2);
  half_t* n1T  = (half_t*)alloc((size_t)128 * 256 * 2);
  half_t* n2T  = (half_t*)alloc((size_t)128 * 128 * 2);
  float*  agg  = (float*)alloc((size_t)N * 128 * 4);
  float*  xacc = (float*)alloc((size_t)N * 42 * 4);
  float*  cr   = (float*)alloc((size_t)N * 4);
  float*  cc   = (float*)alloc((size_t)N * 4);
  float*  pcb  = (float*)alloc((size_t)N * 3 * 4);
  float*  chs  = (float*)alloc((size_t)N * 4);

  hipMemsetAsync(agg,  0, (size_t)N * 128 * 4, stream);
  hipMemsetAsync(xacc, 0, (size_t)N * 42 * 4,  stream);
  hipMemsetAsync(cr,   0, (size_t)N * 4,       stream);
  hipMemsetAsync(cc,   0, (size_t)N * 4,       stream);

  transpose_f16_kernel<<<(256 * 128 + 255) / 256, 256, 0, stream>>>(rad_w, radT, 256, 128);
  transpose_f16_kernel<<<(384 * 128 + 255) / 256, 256, 0, stream>>>(e_w1,  e1T,  384, 128);
  transpose_f16_kernel<<<(128 * 128 + 255) / 256, 256, 0, stream>>>(e_w2,  e2T,  128, 128);
  transpose_f16_kernel<<<(128 * 128 + 255) / 256, 256, 0, stream>>>(c_w1,  c1T,  128, 128);
  transpose_f16_kernel<<<(256 * 128 + 255) / 256, 256, 0, stream>>>(n_w1,  n1T,  256, 128);
  transpose_f16_kernel<<<(128 * 128 + 255) / 256, 256, 0, stream>>>(n_w2,  n2T,  128, 128);

  conv_h_kernel<<<((size_t)N * 128 + 255) / 256, 256, 0, stream>>>(h, hh, N * 128);
  node_prep_kernel<<<(N + 255) / 256, 256, 0, stream>>>(x, cw, pcb, chs, N);
  count_kernel<<<(E + 255) / 256, 256, 0, stream>>>(row, col, cr, cc, E);

  edge_kernel<<<(E + 15) / 16, 256, 0, stream>>>(
      x, cw, attr, row, col, hh,
      radT, rad_b, e1T, e_b1, e2T, e_b2, att_w, att_b,
      c1T, c_b1, c_w2, c_b2, pcb, chs, agg, xacc, E);

  float* outH = (float*)d_out;
  float* outX = outH + (size_t)N * 128;

  node_kernel<<<(N + 15) / 16, 256, 0, stream>>>(
      h, hh, agg, cc, n1T, n_b1, n2T, n_b2, ln_g, ln_b, outH, N);

  xout_kernel<<<((size_t)N * 42 + 255) / 256, 256, 0, stream>>>(x, xacc, cr, outX, N);
}
